// RecurrentBlock_80539226734608
// MI455X (gfx1250) — matrix operand order repaired and checked by reading, not compiled, not run on hardware
//
#include <hip/hip_runtime.h>
#include <cstdint>

typedef __attribute__((ext_vector_type(16))) __bf16 v16bf;
typedef __attribute__((ext_vector_type(8)))  float  v8f;

#define B_   4
#define S_   2048
#define D_   2048
#define H_   2560
#define NCH  16
#define CL   (S_ / NCH)   // 128

// ---------------- helpers ----------------
__device__ __forceinline__ uint16_t f2bf(float f) {
    uint32_t x = __float_as_uint(f);
    uint32_t r = x + 0x7FFFu + ((x >> 16) & 1u);   // round-to-nearest-even
    return (uint16_t)(r >> 16);
}
__device__ __forceinline__ float bf2f(uint16_t b) {
    return __uint_as_float((uint32_t)b << 16);
}
__device__ __forceinline__ float gelu_tanh(float x) {
    float x3 = x * x * x;
    return 0.5f * x * (1.0f + tanhf(0.7978845608028654f * (x + 0.044715f * x3)));
}
__device__ __forceinline__ float sigmoidf(float x) {
    return 1.0f / (1.0f + expf(-x));
}

union BFrag { v16bf v; uint32_t u[8]; uint4 q[2]; };

// ---------------- one-time conversions ----------------
__global__ __launch_bounds__(256) void cvt_bf16(const float* __restrict__ in,
                                                uint16_t* __restrict__ out, size_t n) {
    size_t i = (size_t)blockIdx.x * 256 + threadIdx.x;
    if (i < n) out[i] = f2bf(in[i]);
}

// W [K][N] fp32  ->  Wt [N][K] bf16 (tiled transpose, coalesced both sides)
__global__ __launch_bounds__(256) void transpose_cvt(const float* __restrict__ W,
                                                     uint16_t* __restrict__ Wt,
                                                     int K, int N) {
    __shared__ uint16_t tile[32][33];
    int k0 = blockIdx.y * 32, n0 = blockIdx.x * 32;
    int tx = threadIdx.x, ty = threadIdx.y;          // 32 x 8
    #pragma unroll
    for (int j = 0; j < 4; ++j)
        tile[ty + 8 * j][tx] = f2bf(W[(size_t)(k0 + ty + 8 * j) * N + n0 + tx]);
    __syncthreads();
    #pragma unroll
    for (int j = 0; j < 4; ++j)
        Wt[(size_t)(n0 + ty + 8 * j) * K + k0 + tx] = tile[tx][ty + 8 * j];
}

// ---------------- WMMA GEMM: out[M,N] = act(A[M,K]bf16 * Wt[N,K]bf16^T + bias) ----------
// BM=128 BN=128 BK=32; 256 threads = 8 waves (2x4); wave tile 64x32 = 4x2 wmma tiles.
template <int ACT, bool OUTBF>
__global__ __launch_bounds__(256) void gemm_bf16t(
    const uint16_t* __restrict__ A,   // [M][K] bf16 row-major
    const uint16_t* __restrict__ Wt,  // [N][K] bf16 (pre-transposed weights)
    const float* __restrict__ bias,
    void* __restrict__ outp, int M, int N, int K)
{
    __shared__ uint32_t As[128 * 20];   // 128 rows x 32 bf16 (16 dw) padded to 20 dw
    __shared__ uint32_t Bs[128 * 20];   // 128 cols x 32 bf16, padded to 20 dw

    const int tid  = threadIdx.x;
    const int lane = tid & 31;
    const int wave = tid >> 5;
    const int wm   = wave >> 2;         // 0..1 (M)
    const int wn   = wave & 3;          // 0..3 (N)
    const int hi   = lane >> 4;
    const int ml   = lane & 15;

    const int m0 = blockIdx.y * 128;
    const int n0 = blockIdx.x * 128;

    v8f acc[4][2] = {};

    for (int kk = 0; kk < K; kk += 32) {
        // stage A & B tiles: 128 rows x 32 bf16 each, b128 transfers (2 chunks/thread)
        #pragma unroll
        for (int j = 0; j < 2; ++j) {
            int idx = tid + 256 * j;
            int r = idx >> 2, c = idx & 3;           // c: which 8-bf16 chunk
            uint4 va = *(const uint4*)(A  + (size_t)(m0 + r) * K + kk + c * 8);
            *(uint4*)&As[r * 20 + c * 4] = va;
            uint4 vb = *(const uint4*)(Wt + (size_t)(n0 + r) * K + kk + c * 8);
            *(uint4*)&Bs[r * 20 + c * 4] = vb;
        }
        __syncthreads();

        // fragments (ISA VGPR layouts), all b128 LDS reads, conflict-free (pitch 20 dw)
        BFrag fa[4], fb[2];
        #pragma unroll
        for (int t = 0; t < 4; ++t) {
            int base = (wm * 64 + t * 16 + ml) * 20 + hi * 4;
            fa[t].q[0] = *(const uint4*)&As[base];        // K 0-7 / 8-15
            fa[t].q[1] = *(const uint4*)&As[base + 8];    // K 16-23 / 24-31
        }
        #pragma unroll
        for (int t = 0; t < 2; ++t) {
            int base = (wn * 32 + t * 16 + ml) * 20 + hi * 4;
            fb[t].q[0] = *(const uint4*)&Bs[base];        // K 0-7 / 8-15
            fb[t].q[1] = *(const uint4*)&Bs[base + 8];    // K 16-23 / 24-31
        }
        #pragma unroll
        for (int tm = 0; tm < 4; ++tm)
            #pragma unroll
            for (int tn = 0; tn < 2; ++tn)
                acc[tm][tn] = __builtin_amdgcn_wmma_f32_16x16x32_bf16(
                    false, fa[tm].v, false, fb[tn].v, (short)0, acc[tm][tn], false, false);

        __syncthreads();
    }

    // epilogue: C layout VGPR v -> M = v + 8*hi, N = ml
    #pragma unroll
    for (int tm = 0; tm < 4; ++tm)
        #pragma unroll
        for (int tn = 0; tn < 2; ++tn)
            #pragma unroll
            for (int v = 0; v < 8; ++v) {
                int rg = m0 + wm * 64 + tm * 16 + v + hi * 8;
                int cg = n0 + wn * 32 + tn * 16 + ml;
                float val = acc[tm][tn][v] + bias[cg];
                if (ACT == 1) val = gelu_tanh(val);
                else if (ACT == 2) val = sigmoidf(val);
                if (OUTBF) ((uint16_t*)outp)[(size_t)rg * N + cg] = f2bf(val);
                else       ((float*)outp)[(size_t)rg * N + cg]    = val;
            }
}

// ---------------- causal depthwise conv, kernel=4; writes f32 + bf16 ----------------
__global__ __launch_bounds__(256) void conv_kernel(
    const float* __restrict__ rin, const float* __restrict__ w,
    const float* __restrict__ cb, float* __restrict__ outf,
    uint16_t* __restrict__ outb)
{
    size_t idx = (size_t)blockIdx.x * 256 + threadIdx.x;
    int h = (int)(idx % H_);
    size_t bs = idx / H_;
    int s = (int)(bs % S_);
    int b = (int)(bs / S_);
    float acc = cb[h];
    #pragma unroll
    for (int k = 0; k < 4; ++k) {
        int ss = s - 3 + k;
        if (ss >= 0)
            acc += w[k * H_ + h] * rin[((size_t)b * S_ + ss) * H_ + h];
    }
    outf[idx] = acc;
    outb[idx] = f2bf(acc);
}

// ---------------- gate math: a = exp(-8*softplus(-lam)*r); d = beta*i*u ----------------
__global__ __launch_bounds__(256) void gates_kernel(
    const float* rr, const float* ii, const float* uu,
    const float* __restrict__ lam, float* aa, float* dd)
{
    size_t idx = (size_t)blockIdx.x * 256 + threadIdx.x;
    int h = (int)(idx % H_);
    float nl = -lam[h];
    float sp = (nl > 20.f) ? nl : log1pf(expf(nl));   // softplus(-lam)
    float r = rr[idx], i = ii[idx], u = uu[idx];
    float log_a = -8.0f * sp * r;
    float a = expf(log_a);
    float beta = sqrtf(fmaxf(1.0f - expf(2.0f * log_a), 1e-12f));
    aa[idx] = a;
    dd[idx] = beta * i * u;
}

// ---------------- chunked scan: h_t = a_t*h + d_t ----------------
__global__ __launch_bounds__(256) void scan_p1(
    const float* __restrict__ a, const float* __restrict__ d,
    float* __restrict__ P, float* __restrict__ L)
{
    size_t idx = (size_t)blockIdx.x * 256 + threadIdx.x;   // B*NCH*H threads
    int h = (int)(idx % H_);
    size_t t = idx / H_;
    int c = (int)(t % NCH);
    int b = (int)(t / NCH);
    float p = 1.0f, l = 0.0f;
    size_t base = ((size_t)b * S_ + (size_t)c * CL) * H_ + h;
    for (int s = 0; s < CL; ++s) {
        float av = a[base], dv = d[base];
        l = av * l + dv;
        p *= av;
        base += H_;
    }
    size_t o = ((size_t)b * NCH + c) * H_ + h;
    P[o] = p; L[o] = l;
}

__global__ __launch_bounds__(256) void scan_p2(
    const float* __restrict__ P, const float* __restrict__ L, float* __restrict__ C)
{
    size_t idx = (size_t)blockIdx.x * 256 + threadIdx.x;   // B*H threads
    int h = (int)(idx % H_);
    int b = (int)(idx / H_);
    float carry = 0.0f;
    for (int c = 0; c < NCH; ++c) {
        size_t o = ((size_t)b * NCH + c) * H_ + h;
        C[o] = carry;
        carry = L[o] + P[o] * carry;
    }
}

// fixup + fused product: leftb <- bf16( left * right ), in place
__global__ __launch_bounds__(256) void scan_p3(
    const float* __restrict__ a, const float* __restrict__ d,
    const float* __restrict__ C, uint16_t* leftb)
{
    size_t idx = (size_t)blockIdx.x * 256 + threadIdx.x;   // B*NCH*H threads
    int h = (int)(idx % H_);
    size_t t = idx / H_;
    int c = (int)(t % NCH);
    int b = (int)(t / NCH);
    float hstate = C[((size_t)b * NCH + c) * H_ + h];
    size_t base = ((size_t)b * S_ + (size_t)c * CL) * H_ + h;
    for (int s = 0; s < CL; ++s) {
        hstate = a[base] * hstate + d[base];
        float prod = bf2f(leftb[base]) * hstate;
        leftb[base] = f2bf(prod);
        base += H_;
    }
}

// ---------------- launcher ----------------
extern "C" void kernel_launch(void* const* d_in, const int* in_sizes, int n_in,
                              void* d_out, int out_size, void* d_ws, size_t ws_size,
                              hipStream_t stream) {
    const float* x     = (const float*)d_in[0];
    const float* Wl    = (const float*)d_in[1];
    const float* bl    = (const float*)d_in[2];
    const float* Wr    = (const float*)d_in[3];
    const float* br    = (const float*)d_in[4];
    const float* convw = (const float*)d_in[5];
    const float* convb = (const float*)d_in[6];
    const float* Wa    = (const float*)d_in[7];
    const float* ba    = (const float*)d_in[8];
    const float* Wi    = (const float*)d_in[9];
    const float* bi    = (const float*)d_in[10];
    const float* lam   = (const float*)d_in[11];
    const float* Wo    = (const float*)d_in[12];
    const float* bo    = (const float*)d_in[13];
    float* out = (float*)d_out;

    const int    M   = B_ * S_;                     // 8192
    const size_t BSH = (size_t)B_ * S_ * H_;        // 20,971,520
    const size_t MD  = (size_t)M * D_;              // 16,777,216
    const size_t BNH = (size_t)B_ * NCH * H_;       // 163,840

    // workspace layout (all region sizes are multiples of 16 bytes)
    char* p = (char*)d_ws;
    float* rin  = (float*)p;  p += BSH * 4;         // r_in (f32)
    float* uf   = (float*)p;  p += BSH * 4;         // conv out (f32)
    float* rbuf = (float*)p;  p += BSH * 4;         // r -> a (in place)
    float* ibuf = (float*)p;  p += BSH * 4;         // i -> drive (in place)
    float* Pbuf = (float*)p;  p += BNH * 4;
    float* Lbuf = (float*)p;  p += BNH * 4;
    float* Cbuf = (float*)p;  p += BNH * 4;
    uint16_t* xb    = (uint16_t*)p; p += MD  * 2;   // x bf16
    uint16_t* leftb = (uint16_t*)p; p += BSH * 2;   // left bf16 -> left*right bf16
    uint16_t* ub    = (uint16_t*)p; p += BSH * 2;   // conv out bf16
    uint16_t* Wlt   = (uint16_t*)p; p += (size_t)D_ * H_ * 2;
    uint16_t* Wrt   = (uint16_t*)p; p += (size_t)D_ * H_ * 2;
    uint16_t* Wat   = (uint16_t*)p; p += (size_t)H_ * H_ * 2;
    uint16_t* Wit   = (uint16_t*)p; p += (size_t)H_ * H_ * 2;
    uint16_t* Wot   = (uint16_t*)p; p += (size_t)H_ * D_ * 2;

    dim3 blk(256);
    dim3 tblk(32, 8);

    // one-time precision conversion + weight transposes
    cvt_bf16<<<(unsigned)((MD + 255) / 256), blk, 0, stream>>>(x, xb, MD);
    transpose_cvt<<<dim3(H_ / 32, D_ / 32), tblk, 0, stream>>>(Wl, Wlt, D_, H_);
    transpose_cvt<<<dim3(H_ / 32, D_ / 32), tblk, 0, stream>>>(Wr, Wrt, D_, H_);
    transpose_cvt<<<dim3(H_ / 32, H_ / 32), tblk, 0, stream>>>(Wa, Wat, H_, H_);
    transpose_cvt<<<dim3(H_ / 32, H_ / 32), tblk, 0, stream>>>(Wi, Wit, H_, H_);
    transpose_cvt<<<dim3(D_ / 32, H_ / 32), tblk, 0, stream>>>(Wo, Wot, H_, D_);

    dim3 gH(H_ / 128, M / 128);                     // N = 2560
    dim3 gO(D_ / 128, M / 128);                     // N = 2048

    // input projections
    gemm_bf16t<1, true ><<<gH, blk, 0, stream>>>(xb, Wlt, bl, leftb, M, H_, D_);
    gemm_bf16t<0, false><<<gH, blk, 0, stream>>>(xb, Wrt, br, rin,   M, H_, D_);
    // causal depthwise conv
    conv_kernel<<<(unsigned)(BSH / 256), blk, 0, stream>>>(rin, convw, convb, uf, ub);
    // gate projections (sigmoid fused)
    gemm_bf16t<2, false><<<gH, blk, 0, stream>>>(ub, Wat, ba, rbuf, M, H_, H_);
    gemm_bf16t<2, false><<<gH, blk, 0, stream>>>(ub, Wit, bi, ibuf, M, H_, H_);
    // a / drive (in place over r / i)
    gates_kernel<<<(unsigned)(BSH / 256), blk, 0, stream>>>(rbuf, ibuf, uf, lam, rbuf, ibuf);
    // chunked linear scan; p3 fuses left*right product into leftb (bf16)
    scan_p1<<<(unsigned)(B_ * NCH * H_ / 256), blk, 0, stream>>>(rbuf, ibuf, Pbuf, Lbuf);
    scan_p2<<<(unsigned)(B_ * H_ / 256),       blk, 0, stream>>>(Pbuf, Lbuf, Cbuf);
    scan_p3<<<(unsigned)(B_ * NCH * H_ / 256), blk, 0, stream>>>(rbuf, ibuf, Cbuf, leftb);
    // output projection
    gemm_bf16t<0, false><<<gO, blk, 0, stream>>>(leftb, Wot, bo, out, M, D_, H_);
}
